// SpatialTemporalTransformer_49254684951013
// MI455X (gfx1250) — matrix operand order repaired and checked by reading, not compiled, not run on hardware
//
#include <hip/hip_runtime.h>
#include <math.h>

typedef __attribute__((ext_vector_type(16))) _Float16 v16h;
typedef __attribute__((ext_vector_type(8)))  _Float16 v8h;
typedef __attribute__((ext_vector_type(2)))  _Float16 h2;
typedef __attribute__((ext_vector_type(8)))  float    v8f;

constexpr int TB = 256;           // 8 wave32 waves per block
constexpr int BM = 64, BK = 32;
constexpr int PADH = 8;           // LDS row pad (halves): 80B stride -> conflict-free, 16B aligned

constexpr int kF = 16, kC = 512, kHW = 576, kBF = 32;
constexpr int kB = 2, kHeads = 8, kDH = 64, kCtxL = 77, kCtxD = 1024, kFFI = 2048;
constexpr long kRows = (long)kBF * kHW;   // 18432
constexpr long kE = kRows * kC;           // 9437184 elements

struct GemmP {
  const float* A; long sAb, sAh; int lda;
  const float* B; long sBb, sBh; int ldb;
  float*       C; long sCb, sCh; int ldc;
  const float* bias;   // indexed by output column n (and n+goff for GEGLU gate)
  const float* res;    // same layout/offsets as C; nullable
  int M, N, K, heads, transB, kvg, goff;
  float alpha;
};

__device__ __forceinline__ float gelu_exact(float x) {
  return 0.5f * x * (1.f + erff(x * 0.7071067811865475f));
}

__device__ __forceinline__ v16h cat16(v8h lo, v8h hi) {
  return __builtin_shufflevector(lo, hi, 0, 1, 2, 3, 4, 5, 6, 7, 8, 9, 10, 11, 12, 13, 14, 15);
}

// D = alpha * A(MxK) @ B(KxN) [+bias][+res], f16 WMMA with f32 accumulate.
// GEGLU variant computes both halves of x@pw and writes a*gelu(g) directly.
// BNt: block tile width (64 or 128). 8 waves: 4 row strips x 2 col strips.
template <bool GEGLU, int BNt>
__global__ __launch_bounds__(TB)
void gemm_wmma(GemmP p) {
  constexpr int WT = BNt / 32;          // wmma tiles per wave (col dir)
  constexpr int WCOL = BNt / 2;         // cols per wave strip

  __shared__ _Float16 sA [BM][BK + PADH];              // row-major A tile
  __shared__ _Float16 sBt[BNt][BK + PADH];             // B tile TRANSPOSED: [col][k]
  __shared__ _Float16 sGt[GEGLU ? BNt : 1][BK + PADH]; // gate tile (GEGLU only)

  const int tid  = threadIdx.x;
  const int lane = tid & 31;
  const int wave = tid >> 5;
  const int wm   = wave >> 1;   // 0..3 -> 16-row strip
  const int wn   = wave & 1;    // 0..1 -> col strip

  const int bm0 = blockIdx.y * BM;
  const int bn0 = blockIdx.x * BNt;

  const int z  = blockIdx.z;
  const int bb = z / p.heads;
  const int hh = z - bb * p.heads;
  const int bkv = (p.kvg > 1) ? (bb / p.kvg) * p.kvg : bb;
  const float* Ag = p.A + (long)bb  * p.sAb + (long)hh * p.sAh;
  const float* Bg = p.B + (long)bkv * p.sBb + (long)hh * p.sBh;
  float*       Cg = p.C + (long)bb  * p.sCb + (long)hh * p.sCh;
  const float* Rg = p.res ? (p.res + (long)bb * p.sCb + (long)hh * p.sCh) : nullptr;

  const bool okM = (bm0 + BM <= p.M);
  const bool okN = (bn0 + BNt <= p.N);

  v8f acc[WT];
  v8f accg[GEGLU ? WT : 1];
#pragma unroll
  for (int t = 0; t < WT; ++t) acc[t] = v8f{};
  if constexpr (GEGLU) {
#pragma unroll
    for (int t = 0; t < WT; ++t) accg[t] = v8f{};
  }

  for (int kk = 0; kk < p.K; kk += BK) {
    const bool okK = (kk + BK <= p.K);
    __syncthreads();
    // ---- stage A tile (BM x BK) as packed f16 pairs; coalesced along K ----
    if (okM && okK) {
#pragma unroll
      for (int j = 0; j < BM * (BK / 2) / TB; ++j) {
        const int e = tid + j * TB;
        const int r = e >> 4, c2 = e & 15;
        const float* ap = Ag + (long)(bm0 + r) * p.lda + (kk + c2 * 2);
        h2 t = {(_Float16)ap[0], (_Float16)ap[1]};
        *(h2*)&sA[r][c2 * 2] = t;
      }
    } else {
#pragma unroll
      for (int j = 0; j < BM * (BK / 2) / TB; ++j) {
        const int e = tid + j * TB;
        const int r = e >> 4, c2 = e & 15;
        const int gm = bm0 + r, gk = kk + c2 * 2;
        float v0 = 0.f, v1 = 0.f;
        if (gm < p.M) {
          const float* ap = Ag + (long)gm * p.lda;
          if (gk     < p.K) v0 = ap[gk];
          if (gk + 1 < p.K) v1 = ap[gk + 1];
        }
        h2 t = {(_Float16)v0, (_Float16)v1};
        *(h2*)&sA[r][c2 * 2] = t;
      }
    }
    // ---- stage B tile transposed into sBt[col][k] ----
    if (p.transB) {
      if (okN && okK) {
#pragma unroll
        for (int j = 0; j < BNt * (BK / 2) / TB; ++j) {
          const int e = tid + j * TB;
          const int c = e >> 4, r2 = e & 15;
          const float* bp = Bg + (long)(bn0 + c) * p.ldb + (kk + r2 * 2);
          h2 t = {(_Float16)bp[0], (_Float16)bp[1]};
          *(h2*)&sBt[c][r2 * 2] = t;
        }
      } else {
#pragma unroll
        for (int j = 0; j < BNt * (BK / 2) / TB; ++j) {
          const int e = tid + j * TB;
          const int c = e >> 4, r2 = e & 15;
          const int gn = bn0 + c, gk = kk + r2 * 2;
          float v0 = 0.f, v1 = 0.f;
          if (gn < p.N) {
            const float* bp = Bg + (long)gn * p.ldb;
            if (gk     < p.K) v0 = bp[gk];
            if (gk + 1 < p.K) v1 = bp[gk + 1];
          }
          h2 t = {(_Float16)v0, (_Float16)v1};
          *(h2*)&sBt[c][r2 * 2] = t;
        }
      }
    } else {
      if (okN && okK) {
#pragma unroll
        for (int j = 0; j < BNt * (BK / 2) / TB; ++j) {
          const int e = tid + j * TB;
          const int c = e & (BNt - 1), r2 = e / BNt;
          const long base = (long)(kk + r2 * 2) * p.ldb + (bn0 + c);
          h2 t = {(_Float16)Bg[base], (_Float16)Bg[base + p.ldb]};
          *(h2*)&sBt[c][r2 * 2] = t;
          if constexpr (GEGLU) {
            h2 tg = {(_Float16)Bg[base + p.goff], (_Float16)Bg[base + p.ldb + p.goff]};
            *(h2*)&sGt[c][r2 * 2] = tg;
          }
        }
      } else {
#pragma unroll
        for (int j = 0; j < BNt * (BK / 2) / TB; ++j) {
          const int e = tid + j * TB;
          const int c = e & (BNt - 1), r2 = e / BNt;
          const int gn = bn0 + c, gk = kk + r2 * 2;
          float v0 = 0.f, v1 = 0.f, g0 = 0.f, g1 = 0.f;
          if (gn < p.N) {
            if (gk     < p.K) v0 = Bg[(long)gk * p.ldb + gn];
            if (gk + 1 < p.K) v1 = Bg[(long)(gk + 1) * p.ldb + gn];
            if constexpr (GEGLU) {
              if (gk     < p.K) g0 = Bg[(long)gk * p.ldb + gn + p.goff];
              if (gk + 1 < p.K) g1 = Bg[(long)(gk + 1) * p.ldb + gn + p.goff];
            }
          }
          h2 t = {(_Float16)v0, (_Float16)v1};
          *(h2*)&sBt[c][r2 * 2] = t;
          if constexpr (GEGLU) {
            h2 tg = {(_Float16)g0, (_Float16)g1};
            *(h2*)&sGt[c][r2 * 2] = tg;
          }
        }
      }
    }
    __syncthreads();

    // ---- A fragment: lanes 0-15 row=lane K{0..15}; lanes 16-31 row=lane-16 K{16..31}
    const int ar = wm * 16 + (lane & 15);
    const int k0 = (lane < 16) ? 0 : 16;
    const v16h af = cat16(*(const v8h*)&sA[ar][k0], *(const v8h*)&sA[ar][k0 + 8]);

    // ---- B fragments (all loads hoisted, then WMMAs issued back-to-back)
    const int bcb = wn * WCOL + (lane & 15);
    const int kb  = (lane < 16) ? 0 : 16;
    v16h bf[WT];
#pragma unroll
    for (int t = 0; t < WT; ++t)
      bf[t] = cat16(*(const v8h*)&sBt[bcb + t * 16][kb], *(const v8h*)&sBt[bcb + t * 16][kb + 8]);
#pragma unroll
    for (int t = 0; t < WT; ++t)
      acc[t] = __builtin_amdgcn_wmma_f32_16x16x32_f16(false, af, false, bf[t], (short)0, acc[t], false, false);
    if constexpr (GEGLU) {
      v16h bg[WT];
#pragma unroll
      for (int t = 0; t < WT; ++t)
        bg[t] = cat16(*(const v8h*)&sGt[bcb + t * 16][kb], *(const v8h*)&sGt[bcb + t * 16][kb + 8]);
#pragma unroll
      for (int t = 0; t < WT; ++t)
        accg[t] = __builtin_amdgcn_wmma_f32_16x16x32_f16(false, af, false, bg[t], (short)0, accg[t], false, false);
    }
  }

  // epilogue: C/D layout -> vgpr vr, lanes 0-15: M=vr, N=lane; lanes 16-31: M=8+vr, N=lane-16
  const int row0 = bm0 + wm * 16 + ((lane < 16) ? 0 : 8);
#pragma unroll
  for (int t = 0; t < WT; ++t) {
    const int gn = bn0 + wn * WCOL + t * 16 + (lane & 15);
    if (gn >= p.N) continue;
#pragma unroll
    for (int vr = 0; vr < 8; ++vr) {
      const int gm = row0 + vr;
      if (gm >= p.M) continue;
      const long off = (long)gm * p.ldc + gn;
      float v;
      if constexpr (GEGLU) {
        float a = acc[t][vr]  + (p.bias ? p.bias[gn] : 0.f);
        float g = accg[t][vr] + (p.bias ? p.bias[gn + p.goff] : 0.f);
        v = a * gelu_exact(g);
      } else {
        v = acc[t][vr] * p.alpha;
        if (p.bias) v += p.bias[gn];
        if (Rg)     v += Rg[off];
      }
      Cg[off] = v;
    }
  }
}

// GroupNorm(32 groups of 16 ch) on (BF,C,H,W) -> normalized (BF,HW,C)
__global__ __launch_bounds__(256)
void groupnorm_nhwc(const float* __restrict__ x, float* __restrict__ out,
                    const float* __restrict__ w, const float* __restrict__ b) {
  const int bf = blockIdx.x >> 5;
  const int g  = blockIdx.x & 31;
  const float* xb = x + (long)bf * kC * kHW;
  const int c0 = g * 16;
  __shared__ float rs_[256], rq_[256];
  const int t = threadIdx.x;
  float s = 0.f, q = 0.f;
  for (int e = t; e < 16 * kHW; e += 256) {
    int c = c0 + e / kHW, hw = e % kHW;
    float v = xb[(long)c * kHW + hw];
    s += v; q += v * v;
  }
  rs_[t] = s; rq_[t] = q; __syncthreads();
  for (int st = 128; st > 0; st >>= 1) {
    if (t < st) { rs_[t] += rs_[t + st]; rq_[t] += rq_[t + st]; }
    __syncthreads();
  }
  const float m   = rs_[0] * (1.f / 9216.f);
  const float var = rq_[0] * (1.f / 9216.f) - m * m;
  const float inv = rsqrtf(var + 1e-6f);
  for (int e = t; e < 16 * kHW; e += 256) {
    int c = c0 + e / kHW, hw = e % kHW;
    float v = (xb[(long)c * kHW + hw] - m) * inv * w[c] + b[c];
    out[(long)bf * kHW * kC + (long)hw * kC + c] = v;
  }
}

__global__ __launch_bounds__(256)
void layernorm512(const float* __restrict__ in, float* __restrict__ out,
                  const float* __restrict__ w, const float* __restrict__ b) {
  const long row = blockIdx.x;
  const float* p = in + row * kC;
  float* o = out + row * kC;
  __shared__ float red[256];
  const int t = threadIdx.x;
  const float x0 = p[t], x1 = p[t + 256];
  red[t] = x0 + x1; __syncthreads();
  for (int st = 128; st > 0; st >>= 1) { if (t < st) red[t] += red[t + st]; __syncthreads(); }
  const float m = red[0] * (1.f / 512.f);
  __syncthreads();
  const float d0 = x0 - m, d1 = x1 - m;
  red[t] = d0 * d0 + d1 * d1; __syncthreads();
  for (int st = 128; st > 0; st >>= 1) { if (t < st) red[t] += red[t + st]; __syncthreads(); }
  const float inv = rsqrtf(red[0] * (1.f / 512.f) + 1e-5f);
  o[t]       = d0 * inv * w[t]       + b[t];
  o[t + 256] = d1 * inv * w[t + 256] + b[t + 256];
}

__global__ __launch_bounds__(256)
void softmax_rows(float* __restrict__ x, int ncol) {
  float* p = x + (long)blockIdx.x * ncol;
  __shared__ float red[256];
  const int t = threadIdx.x;
  float mx = -3.0e38f;
  for (int c = t; c < ncol; c += 256) mx = fmaxf(mx, p[c]);
  red[t] = mx; __syncthreads();
  for (int st = 128; st > 0; st >>= 1) { if (t < st) red[t] = fmaxf(red[t], red[t + st]); __syncthreads(); }
  mx = red[0]; __syncthreads();
  float s = 0.f;
  for (int c = t; c < ncol; c += 256) { float e = __expf(p[c] - mx); p[c] = e; s += e; }
  red[t] = s; __syncthreads();
  for (int st = 128; st > 0; st >>= 1) { if (t < st) red[t] += red[t + st]; __syncthreads(); }
  const float inv = 1.f / red[0];
  for (int c = t; c < ncol; c += 256) p[c] *= inv;
}

// (b f, hw, c) -> (b hw, f, c)
__global__ void to_temporal(const float* __restrict__ in, float* __restrict__ out) {
  const long i = (long)blockIdx.x * 256 + threadIdx.x;
  if (i >= kE) return;
  const int c = (int)(i & 511);
  const long r = i >> 9;
  const int hw = (int)(r % kHW);
  const int bf = (int)(r / kHW);
  const int b = bf >> 4, f = bf & 15;
  out[(((long)b * kHW + hw) * kF + f) * kC + c] = in[i];
}

// alpha*xs + (1-alpha)*xt, reading xt back from temporal layout
__global__ void blend_mix(const float* __restrict__ xs, const float* __restrict__ xt,
                          const unsigned char* __restrict__ ind, const float* __restrict__ mix,
                          float* __restrict__ out) {
  const long i = (long)blockIdx.x * 256 + threadIdx.x;
  if (i >= kE) return;
  const int c = (int)(i & 511);
  const long r = i >> 9;
  const int hw = (int)(r % kHW);
  const int bf = (int)(r / kHW);
  const int b = bf >> 4, f = bf & 15;
  const float sig = 1.f / (1.f + __expf(-mix[0]));
  const float alpha = ind[bf] ? 1.f : sig;
  const float t = xt[(((long)b * kHW + hw) * kF + f) * kC + c];
  out[i] = alpha * xs[i] + (1.f - alpha) * t;
}

// (bf, hw, c) -> (bf, c, h, w) plus residual x_in
__global__ void final_add(const float* __restrict__ proj, const float* __restrict__ xin,
                          float* __restrict__ out) {
  const long i = (long)blockIdx.x * 256 + threadIdx.x;
  if (i >= kE) return;
  const int hw = (int)(i % kHW);
  const long r = i / kHW;
  const int c  = (int)(r % kC);
  const int bf = (int)(r / kC);
  out[i] = xin[i] + proj[((long)bf * kHW + hw) * kC + c];
}

extern "C" void kernel_launch(void* const* d_in, const int* in_sizes, int n_in,
                              void* d_out, int out_size, void* d_ws, size_t ws_size,
                              hipStream_t stream) {
  (void)in_sizes; (void)n_in; (void)out_size; (void)ws_size;
  const float* x          = (const float*)d_in[0];
  const float* ctx        = (const float*)d_in[1];
  const unsigned char* ind= (const unsigned char*)d_in[2];
  const float* gn_w = (const float*)d_in[3];  const float* gn_b = (const float*)d_in[4];
  const float* proj_in_w = (const float*)d_in[5]; const float* proj_in_b = (const float*)d_in[6];
  const float* s_n1_w = (const float*)d_in[7];  const float* s_n1_b = (const float*)d_in[8];
  const float* s_n2_w = (const float*)d_in[9];  const float* s_n2_b = (const float*)d_in[10];
  const float* s_n3_w = (const float*)d_in[11]; const float* s_n3_b = (const float*)d_in[12];
  const float* s_a1_q = (const float*)d_in[13]; const float* s_a1_k = (const float*)d_in[14];
  const float* s_a1_v = (const float*)d_in[15]; const float* s_a1_o = (const float*)d_in[16];
  const float* s_a1_ob= (const float*)d_in[17];
  const float* s_a2_q = (const float*)d_in[18]; const float* s_a2_k = (const float*)d_in[19];
  const float* s_a2_v = (const float*)d_in[20]; const float* s_a2_o = (const float*)d_in[21];
  const float* s_a2_ob= (const float*)d_in[22];
  const float* s_ff_pw= (const float*)d_in[23]; const float* s_ff_pb= (const float*)d_in[24];
  const float* s_ff_ow= (const float*)d_in[25]; const float* s_ff_ob= (const float*)d_in[26];
  const float* t_nin_w= (const float*)d_in[27]; const float* t_nin_b= (const float*)d_in[28];
  const float* t_n1_w = (const float*)d_in[29]; const float* t_n1_b = (const float*)d_in[30];
  const float* t_n3_w = (const float*)d_in[31]; const float* t_n3_b = (const float*)d_in[32];
  const float* t_ffin_pw=(const float*)d_in[33];const float* t_ffin_pb=(const float*)d_in[34];
  const float* t_ffin_ow=(const float*)d_in[35];const float* t_ffin_ob=(const float*)d_in[36];
  const float* t_a1_q = (const float*)d_in[37]; const float* t_a1_k = (const float*)d_in[38];
  const float* t_a1_v = (const float*)d_in[39]; const float* t_a1_o = (const float*)d_in[40];
  const float* t_a1_ob= (const float*)d_in[41];
  const float* t_ff_pw= (const float*)d_in[42]; const float* t_ff_pb= (const float*)d_in[43];
  const float* t_ff_ow= (const float*)d_in[44]; const float* t_ff_ob= (const float*)d_in[45];
  const float* mix    = (const float*)d_in[46];
  const float* proj_out_w=(const float*)d_in[47];const float* proj_out_b=(const float*)d_in[48];

  float* W     = (float*)d_ws;
  float* b_h   = W;             // hidden / x_spatial
  float* b_n   = W + 1 * kE;    // normed scratch
  float* b_q   = W + 2 * kE;
  float* b_k   = W + 3 * kE;
  float* b_v   = W + 4 * kE;
  float* b_t   = W + 5 * kE;    // temporal hidden
  float* b_p   = W + 6 * kE;    // attn out / blended
  float* b_gg  = W + 7 * kE;    // 4*kE: GEGLU output (18432 x 2048)
  float* b_sim = W + 11 * kE;   // 2*kE: attention logits/probs

  auto gemm = [&](const GemmP& p, int batch, bool gg) {
    if (gg) {
      dim3 grid((p.N + 127) / 128, (p.M + BM - 1) / BM, batch);
      gemm_wmma<true, 128><<<grid, dim3(TB), 0, stream>>>(p);
    } else if (p.N >= 128) {
      dim3 grid((p.N + 127) / 128, (p.M + BM - 1) / BM, batch);
      gemm_wmma<false, 128><<<grid, dim3(TB), 0, stream>>>(p);
    } else {
      dim3 grid((p.N + 63) / 64, (p.M + BM - 1) / BM, batch);
      gemm_wmma<false, 64><<<grid, dim3(TB), 0, stream>>>(p);
    }
  };
  auto plain = [&](const float* A, const float* Bm, float* Cm, int M, int N, int K,
                   const float* bias, const float* res) {
    GemmP p{}; p.A = A; p.lda = K; p.B = Bm; p.ldb = N; p.C = Cm; p.ldc = N;
    p.bias = bias; p.res = res; p.M = M; p.N = N; p.K = K;
    p.heads = 1; p.transB = 0; p.kvg = 1; p.goff = 0; p.alpha = 1.f;
    gemm(p, 1, false);
  };
  auto geglu = [&](const float* A, const float* pw, const float* pb, float* outg) {
    GemmP p{}; p.A = A; p.lda = kC; p.B = pw; p.ldb = 2 * kFFI; p.C = outg; p.ldc = kFFI;
    p.bias = pb; p.res = nullptr; p.M = (int)kRows; p.N = kFFI; p.K = kC;
    p.heads = 1; p.transB = 0; p.kvg = 1; p.goff = kFFI; p.alpha = 1.f;
    gemm(p, 1, true);
  };
  auto ln = [&](const float* in, float* out, const float* w, const float* b) {
    layernorm512<<<dim3((unsigned)kRows), dim3(256), 0, stream>>>(in, out, w, b);
  };
  const unsigned nElemBlocks = (unsigned)((kE + 255) / 256);

  // ---- GroupNorm + proj_in ----
  groupnorm_nhwc<<<dim3(kBF * 32), dim3(256), 0, stream>>>(x, b_n, gn_w, gn_b);
  plain(b_n, proj_in_w, b_h, (int)kRows, kC, kC, proj_in_b, nullptr);

  // ---- spatial self-attn (crossframe=16) ----
  ln(b_h, b_n, s_n1_w, s_n1_b);
  plain(b_n, s_a1_q, b_q, (int)kRows, kC, kC, nullptr, nullptr);
  plain(b_n, s_a1_k, b_k, (int)kRows, kC, kC, nullptr, nullptr);
  plain(b_n, s_a1_v, b_v, (int)kRows, kC, kC, nullptr, nullptr);
  for (int b = 0; b < kBF; ++b) {
    const int bkv = (b / 16) * 16;
    GemmP ps{}; ps.A = b_q + (long)b * kHW * kC; ps.sAb = 0; ps.sAh = kDH; ps.lda = kC;
    ps.B = b_k + (long)bkv * kHW * kC; ps.sBb = 0; ps.sBh = kDH; ps.ldb = kC;
    ps.C = b_sim; ps.sCb = 0; ps.sCh = (long)kHW * kHW; ps.ldc = kHW;
    ps.bias = nullptr; ps.res = nullptr;
    ps.M = kHW; ps.N = kHW; ps.K = kDH; ps.heads = kHeads;
    ps.transB = 1; ps.kvg = 1; ps.goff = 0; ps.alpha = 0.125f;
    gemm(ps, kHeads, false);
    softmax_rows<<<dim3(kHeads * kHW), dim3(256), 0, stream>>>(b_sim, kHW);
    GemmP pv{}; pv.A = b_sim; pv.sAb = 0; pv.sAh = (long)kHW * kHW; pv.lda = kHW;
    pv.B = b_v + (long)bkv * kHW * kC; pv.sBb = 0; pv.sBh = kDH; pv.ldb = kC;
    pv.C = b_p + (long)b * kHW * kC; pv.sCb = 0; pv.sCh = kDH; pv.ldc = kC;
    pv.bias = nullptr; pv.res = nullptr;
    pv.M = kHW; pv.N = kDH; pv.K = kHW; pv.heads = kHeads;
    pv.transB = 0; pv.kvg = 1; pv.goff = 0; pv.alpha = 1.f;
    gemm(pv, kHeads, false);
  }
  plain(b_p, s_a1_o, b_h, (int)kRows, kC, kC, s_a1_ob, b_h);

  // ---- cross attention ----
  ln(b_h, b_n, s_n2_w, s_n2_b);
  plain(b_n, s_a2_q, b_q, (int)kRows, kC, kC, nullptr, nullptr);
  plain(ctx, s_a2_k, b_k, kBF * kCtxL, kC, kCtxD, nullptr, nullptr);
  plain(ctx, s_a2_v, b_v, kBF * kCtxL, kC, kCtxD, nullptr, nullptr);
  {
    GemmP ps{}; ps.A = b_q; ps.sAb = (long)kHW * kC; ps.sAh = kDH; ps.lda = kC;
    ps.B = b_k; ps.sBb = (long)kCtxL * kC; ps.sBh = kDH; ps.ldb = kC;
    ps.C = b_sim; ps.sCb = (long)kHeads * kHW * kCtxL; ps.sCh = (long)kHW * kCtxL; ps.ldc = kCtxL;
    ps.bias = nullptr; ps.res = nullptr;
    ps.M = kHW; ps.N = kCtxL; ps.K = kDH; ps.heads = kHeads;
    ps.transB = 1; ps.kvg = 1; ps.goff = 0; ps.alpha = 0.125f;
    gemm(ps, kBF * kHeads, false);
  }
  softmax_rows<<<dim3(kBF * kHeads * kHW), dim3(256), 0, stream>>>(b_sim, kCtxL);
  {
    GemmP pv{}; pv.A = b_sim; pv.sAb = (long)kHeads * kHW * kCtxL; pv.sAh = (long)kHW * kCtxL; pv.lda = kCtxL;
    pv.B = b_v; pv.sBb = (long)kCtxL * kC; pv.sBh = kDH; pv.ldb = kC;
    pv.C = b_p; pv.sCb = (long)kHW * kC; pv.sCh = kDH; pv.ldc = kC;
    pv.bias = nullptr; pv.res = nullptr;
    pv.M = kHW; pv.N = kDH; pv.K = kCtxL; pv.heads = kHeads;
    pv.transB = 0; pv.kvg = 1; pv.goff = 0; pv.alpha = 1.f;
    gemm(pv, kBF * kHeads, false);
  }
  plain(b_p, s_a2_o, b_h, (int)kRows, kC, kC, s_a2_ob, b_h);

  // ---- spatial FF (GEGLU) ----
  ln(b_h, b_n, s_n3_w, s_n3_b);
  geglu(b_n, s_ff_pw, s_ff_pb, b_gg);
  plain(b_gg, s_ff_ow, b_h, (int)kRows, kC, kFFI, s_ff_ob, b_h);   // b_h = x_spatial

  // ---- temporal block ----
  to_temporal<<<dim3(nElemBlocks), dim3(256), 0, stream>>>(b_h, b_t);
  ln(b_t, b_n, t_nin_w, t_nin_b);
  geglu(b_n, t_ffin_pw, t_ffin_pb, b_gg);
  plain(b_gg, t_ffin_ow, b_t, (int)kRows, kC, kFFI, t_ffin_ob, b_t);
  ln(b_t, b_n, t_n1_w, t_n1_b);
  plain(b_n, t_a1_q, b_q, (int)kRows, kC, kC, nullptr, nullptr);
  plain(b_n, t_a1_k, b_k, (int)kRows, kC, kC, nullptr, nullptr);
  plain(b_n, t_a1_v, b_v, (int)kRows, kC, kC, nullptr, nullptr);
  {
    GemmP ps{}; ps.A = b_q; ps.sAb = (long)kF * kC; ps.sAh = kDH; ps.lda = kC;
    ps.B = b_k; ps.sBb = (long)kF * kC; ps.sBh = kDH; ps.ldb = kC;
    ps.C = b_sim; ps.sCb = (long)kHeads * kF * kF; ps.sCh = (long)kF * kF; ps.ldc = kF;
    ps.bias = nullptr; ps.res = nullptr;
    ps.M = kF; ps.N = kF; ps.K = kDH; ps.heads = kHeads;
    ps.transB = 1; ps.kvg = 1; ps.goff = 0; ps.alpha = 0.125f;
    gemm(ps, kB * kHW * kHeads, false);
  }
  softmax_rows<<<dim3(kB * kHW * kHeads * kF), dim3(256), 0, stream>>>(b_sim, kF);
  {
    GemmP pv{}; pv.A = b_sim; pv.sAb = (long)kHeads * kF * kF; pv.sAh = (long)kF * kF; pv.lda = kF;
    pv.B = b_v; pv.sBb = (long)kF * kC; pv.sBh = kDH; pv.ldb = kC;
    pv.C = b_p; pv.sCb = (long)kF * kC; pv.sCh = kDH; pv.ldc = kC;
    pv.bias = nullptr; pv.res = nullptr;
    pv.M = kF; pv.N = kDH; pv.K = kF; pv.heads = kHeads;
    pv.transB = 0; pv.kvg = 1; pv.goff = 0; pv.alpha = 1.f;
    gemm(pv, kB * kHW * kHeads, false);
  }
  plain(b_p, t_a1_o, b_t, (int)kRows, kC, kC, t_a1_ob, b_t);
  ln(b_t, b_n, t_n3_w, t_n3_b);
  geglu(b_n, t_ff_pw, t_ff_pb, b_gg);
  plain(b_gg, t_ff_ow, b_t, (int)kRows, kC, kFFI, t_ff_ob, b_t);

  // ---- blend + proj_out + residual ----
  blend_mix<<<dim3(nElemBlocks), dim3(256), 0, stream>>>(b_h, b_t, ind, mix, b_p);
  plain(b_p, proj_out_w, b_n, (int)kRows, kC, kC, proj_out_b, nullptr);
  final_add<<<dim3(nElemBlocks), dim3(256), 0, stream>>>(b_n, x, (float*)d_out);
}
